// MDGCF_42279658062471
// MI455X (gfx1250) — matrix operand order repaired and checked by reading, not compiled, not run on hardware
//
#include <hip/hip_runtime.h>
#include <hip/hip_bf16.h>

// MDGCF on MI455X (gfx1250, wave32, WMMA).
//
// Roofline: dominant cost is streaming the 8192x8192 interaction matrix.
// We materialize a (and a^T) in FP16 (128 MB each) so the six propagation
// GEMMs run on v_wmma_f32_16x16x32_f16 and total HBM traffic is ~1.6 GB
// (~70us at 23.3 TB/s). All GEMMs are phrased as C = A * Bsrc^T with both
// operands row-major so every WMMA fragment is a pair of contiguous
// 16-byte loads per lane (matches CDNA5 16-bit A/B VGPR layouts).

typedef __attribute__((ext_vector_type(16))) _Float16 v16h;
typedef __attribute__((ext_vector_type(8)))  _Float16 v8h;
typedef __attribute__((ext_vector_type(8)))  float    v8f;

#define NROWS 8192
#define DDIM  64

// ---------------- WMMA fragment loaders (v_wmma_f32_16x16x32_f16) ----------
// A fragment: 16x32 tile, row-major source, leading dim ld (elements).
// Lane layout (ISA 7.12.2, 16-bit A 16x32): lane<16 -> row=lane, K {0..7,16..23};
// lane>=16 -> row=lane-16, K {8..15,24..31}.
__device__ __forceinline__ v16h load_fragA(const _Float16* __restrict__ tile, int ld) {
  int lane = threadIdx.x & 31;
  int r = lane & 15, hs = lane >> 4;
  const _Float16* p = tile + (size_t)r * ld + hs * 8;
  v8h lo = *(const v8h*)p;          // K = 8*hs + 0..7
  v8h hi = *(const v8h*)(p + 16);   // K = 8*hs + 16..23
  v16h f;
#pragma unroll
  for (int i = 0; i < 8; ++i) { f[i] = lo[i]; f[8 + i] = hi[i]; }
  return f;
}

// B fragment for C = A * Bsrc^T: Bsrc is N x K row-major; B[k][n] = Bsrc[n][k].
// Lane layout (32x16 B): lane<16 -> col=lane, K {0..7,16..23}; lane>=16 -> col=lane-16, K {8..15,24..31}.
// Per lane this is Bsrc[n][8*hs .. 8*hs+7] and Bsrc[n][8*hs+16 .. 8*hs+23].
__device__ __forceinline__ v16h load_fragB(const _Float16* __restrict__ tile, int ld) {
  int lane = threadIdx.x & 31;
  int n = lane & 15, hs = lane >> 4;
  const _Float16* p = tile + (size_t)n * ld + hs * 8;
  v8h lo = *(const v8h*)p;
  v8h hi = *(const v8h*)(p + 16);
  v16h f;
#pragma unroll
  for (int i = 0; i < 8; ++i) { f[i] = lo[i]; f[8 + i] = hi[i]; }
  return f;
}

__device__ __forceinline__ v8f wmma_f16(v16h a, v16h b, v8f c) {
  return __builtin_amdgcn_wmma_f32_16x16x32_f16(false, a, false, b, (short)0, c,
                                                false, false);
}

// ---------------- init: f32 emb -> f16 copy, f16 transpose, f32 sum --------
__global__ void init_side_kernel(const float* __restrict__ emb,
                                 _Float16* __restrict__ e16,
                                 _Float16* __restrict__ eT,
                                 float* __restrict__ sum) {
  int idx = blockIdx.x * blockDim.x + threadIdx.x;
  if (idx >= NROWS * DDIM) return;
  int r = idx >> 6, d = idx & 63;
  float v = emb[idx];
  sum[idx] = v;                       // layer-0 contribution to mean
  _Float16 h = (_Float16)v;
  e16[idx] = h;
  eT[(size_t)d * NROWS + r] = h;
}

// ---------------- a = adj * (0.5*sigmoid(eu @ ei^T) + 0.5), plus a^T -------
__global__ void compute_a_kernel(const _Float16* __restrict__ eu,
                                 const _Float16* __restrict__ ei,
                                 const float* __restrict__ adj,
                                 _Float16* __restrict__ a,
                                 _Float16* __restrict__ at) {
  int wid = (blockIdx.x * blockDim.x + threadIdx.x) >> 5;  // 0..262143
  int tm = wid >> 9, tn = wid & 511;                       // 512x512 tiles
  int lane = threadIdx.x & 31;
  const _Float16* ta = eu + (size_t)tm * 16 * DDIM;
  const _Float16* tb = ei + (size_t)tn * 16 * DDIM;
  v8f c = {};
  c = wmma_f16(load_fragA(ta, DDIM),      load_fragB(tb, DDIM),      c);
  c = wmma_f16(load_fragA(ta + 32, DDIM), load_fragB(tb + 32, DDIM), c);
  int hs = lane >> 4, n = lane & 15;
  int col = tn * 16 + n;
#pragma unroll
  for (int r = 0; r < 8; ++r) {
    int row = tm * 16 + r + 8 * hs;     // C layout: VGPR r -> M=r (+8 for hi lanes)
    float sim = 1.0f / (1.0f + __expf(-c[r]));
    float ad  = adj[(size_t)row * NROWS + col];
    _Float16 h = (_Float16)(ad * (0.5f * sim + 0.5f));
    a [(size_t)row * NROWS + col] = h;
    at[(size_t)col * NROWS + row] = h;
  }
}

// ---------------- layer GEMM: C[8192x64] = A[8192x8192] * BT[64x8192]^T ----
// One wave owns a 16-row strip and all 64 output columns: A streamed once.
__global__ void layer_gemm_kernel(const _Float16* __restrict__ A,
                                  const _Float16* __restrict__ BT,
                                  float* __restrict__ C) {
  int wid = (blockIdx.x * blockDim.x + threadIdx.x) >> 5;  // 0..511
  int lane = threadIdx.x & 31;
  const _Float16* arow = A + (size_t)wid * 16 * NROWS;
  v8f acc0 = {}, acc1 = {}, acc2 = {}, acc3 = {};
  for (int k0 = 0; k0 < NROWS; k0 += 32) {
    v16h af = load_fragA(arow + k0, NROWS);
    acc0 = wmma_f16(af, load_fragB(BT + k0, NROWS), acc0);
    acc1 = wmma_f16(af, load_fragB(BT + (size_t)16 * NROWS + k0, NROWS), acc1);
    acc2 = wmma_f16(af, load_fragB(BT + (size_t)32 * NROWS + k0, NROWS), acc2);
    acc3 = wmma_f16(af, load_fragB(BT + (size_t)48 * NROWS + k0, NROWS), acc3);
  }
  int hs = lane >> 4, n = lane & 15;
#pragma unroll
  for (int r = 0; r < 8; ++r) {
    int row = wid * 16 + r + 8 * hs;
    float* crow = C + (size_t)row * DDIM + n;
    crow[0]  = acc0[r];
    crow[16] = acc1[r];
    crow[32] = acc2[r];
    crow[48] = acc3[r];
  }
}

// ---------------- postlayer: sum += new; newT(f16) for next layer ----------
__global__ void postlayer_kernel(const float* __restrict__ nw,
                                 float* __restrict__ sum,
                                 _Float16* __restrict__ eT) {
  int idx = blockIdx.x * blockDim.x + threadIdx.x;
  if (idx >= NROWS * DDIM) return;
  int r = idx >> 6, d = idx & 63;
  float v = nw[idx];
  sum[idx] += v;
  eT[(size_t)d * NROWS + r] = (_Float16)v;
}

// ---------------- deterministic global mean (two fixed-order stages) -------
__global__ void reduce1_kernel(const float* __restrict__ x,
                               float* __restrict__ part, int n) {
  __shared__ float sm[256];
  float acc = 0.f;
  for (int i = blockIdx.x * blockDim.x + threadIdx.x; i < n;
       i += blockDim.x * gridDim.x)
    acc += x[i];
  sm[threadIdx.x] = acc;
  __syncthreads();
  for (int st = 128; st > 0; st >>= 1) {
    if ((int)threadIdx.x < st) sm[threadIdx.x] += sm[threadIdx.x + st];
    __syncthreads();
  }
  if (threadIdx.x == 0) part[blockIdx.x] = sm[0];
}

__global__ void reduce2_kernel(const float* __restrict__ part,
                               float* __restrict__ outm, int n, float scale) {
  __shared__ float sm[256];
  float acc = 0.f;
  for (int i = threadIdx.x; i < n; i += 256) acc += part[i];
  sm[threadIdx.x] = acc;
  __syncthreads();
  for (int st = 128; st > 0; st >>= 1) {
    if ((int)threadIdx.x < st) sm[threadIdx.x] += sm[threadIdx.x + st];
    __syncthreads();
  }
  if (threadIdx.x == 0) outm[0] = sm[0] * scale;
}

// ---------------- center rows, per-row inverse norm ------------------------
__global__ void center_norm_kernel(const float* __restrict__ x,
                                   const float* __restrict__ meanp,
                                   float* __restrict__ xc,
                                   float* __restrict__ inv) {
  int wid = (blockIdx.x * blockDim.x + threadIdx.x) >> 5;  // row, 0..8191
  int lane = threadIdx.x & 31;
  float m = meanp[0];
  size_t base = (size_t)wid * DDIM;
  float a0 = x[base + lane] - m;
  float a1 = x[base + 32 + lane] - m;
  xc[base + lane] = a0;
  xc[base + 32 + lane] = a1;
  float ss = a0 * a0 + a1 * a1;
#pragma unroll
  for (int off = 16; off > 0; off >>= 1) ss += __shfl_xor(ss, off, 32);
  if (lane == 0) inv[wid] = 1.0f / (sqrtf(ss) + 1e-8f);
}

// ---------------- per-row top-11 cosine + weighted gather + final mix ------
// One wave per row. Per-lane sorted top-11 over a strided candidate subset,
// then an 11-round wave-wide argmax merge (deterministic lane tie-break),
// drop the largest (self-loop), gather original embeddings, emit output.
__global__ void topk_final_kernel(const float* __restrict__ xc,
                                  const float* __restrict__ inv,
                                  const float* __restrict__ emb,
                                  const float* __restrict__ summ,
                                  float* __restrict__ outp) {
  int wid = (blockIdx.x * blockDim.x + threadIdx.x) >> 5;  // row
  int lane = threadIdx.x & 31;
  int r = wid;

  float xr[DDIM];
  const float* xrp = xc + (size_t)r * DDIM;   // wave-uniform -> scalar loads
#pragma unroll
  for (int k = 0; k < DDIM; ++k) xr[k] = xrp[k];
  float invr = inv[r];

  const float NEG = -3.0e38f;
  float tv[11]; int ti[11];
#pragma unroll
  for (int q = 0; q < 11; ++q) { tv[q] = NEG; ti[q] = -1; }

  for (int cnd = lane; cnd < NROWS; cnd += 32) {
    const float* xcc = xc + (size_t)cnd * DDIM;
    float dot = 0.f;
#pragma unroll
    for (int k = 0; k < DDIM; ++k) dot += xr[k] * xcc[k];
    float s = dot * invr * inv[cnd];
    if (s > tv[10]) {                 // shift-insert, compile-time indices only
      float cv = s; int ci = cnd;
#pragma unroll
      for (int q = 0; q < 11; ++q) {
        if (cv > tv[q]) {
          float t0 = tv[q]; int t1 = ti[q];
          tv[q] = cv; ti[q] = ci; cv = t0; ci = t1;
        }
      }
    }
  }

  float resv[11]; int resi[11];
#pragma unroll
  for (int t = 0; t < 11; ++t) {
    float bv = tv[0]; int bl = lane;
#pragma unroll
    for (int off = 16; off > 0; off >>= 1) {
      float ov = __shfl_xor(bv, off, 32);
      int   ol = __shfl_xor(bl, off, 32);
      if (ov > bv || (ov == bv && ol < bl)) { bv = ov; bl = ol; }
    }
    int bidx = __shfl(ti[0], bl, 32);
    resv[t] = bv; resi[t] = bidx;
    if (lane == bl) {                 // winner pops its head
#pragma unroll
      for (int q = 0; q < 10; ++q) { tv[q] = tv[q + 1]; ti[q] = ti[q + 1]; }
      tv[10] = NEG; ti[10] = -1;
    }
  }

  // Drop t=0 (self-loop); w = vals/TOP_H; gather ORIGINAL embeddings.
  int d0 = lane * 2;
  float s0 = 0.f, s1 = 0.f;
#pragma unroll
  for (int t = 1; t < 11; ++t) {
    float w = resv[t] * 0.1f;
    const float* e = emb + (size_t)resi[t] * DDIM + d0;
    s0 += w * e[0];
    s1 += w * e[1];
  }
  size_t ob = (size_t)r * DDIM + d0;
  outp[ob]     = summ[ob]     * 0.25f + 0.2f * s0;   // mean of 4 + ALPHA*sim_emb
  outp[ob + 1] = summ[ob + 1] * 0.25f + 0.2f * s1;
}

// ---------------------------------------------------------------------------
extern "C" void kernel_launch(void* const* d_in, const int* in_sizes, int n_in,
                              void* d_out, int out_size, void* d_ws, size_t ws_size,
                              hipStream_t stream) {
  const float* emb_u = (const float*)d_in[0];
  const float* emb_i = (const float*)d_in[1];
  const float* adj   = (const float*)d_in[2];
  float* out_users = (float*)d_out;
  float* out_items = out_users + (size_t)NROWS * DDIM;

  char* ws = (char*)d_ws;
  size_t off = 0;
  auto alloc = [&](size_t bytes) -> char* {
    char* p = ws + off;
    off = (off + bytes + 255) & ~(size_t)255;
    return p;
  };

  const size_t EMB_F32 = (size_t)NROWS * DDIM * sizeof(float);   // 2 MB
  const size_t EMB_F16 = (size_t)NROWS * DDIM * 2;               // 1 MB
  const size_t MAT_F16 = (size_t)NROWS * NROWS * 2;              // 128 MB

  _Float16* a_h  = (_Float16*)alloc(MAT_F16);
  _Float16* at_h = (_Float16*)alloc(MAT_F16);
  _Float16* eu16 = (_Float16*)alloc(EMB_F16);
  _Float16* ei16 = (_Float16*)alloc(EMB_F16);
  _Float16* uT   = (_Float16*)alloc(EMB_F16);   // 64 x 8192, f16
  _Float16* iT   = (_Float16*)alloc(EMB_F16);
  float* u_new = (float*)alloc(EMB_F32);
  float* i_new = (float*)alloc(EMB_F32);
  float* u_sum = (float*)alloc(EMB_F32);
  float* i_sum = (float*)alloc(EMB_F32);
  float* xc_u  = (float*)alloc(EMB_F32);
  float* xc_i  = (float*)alloc(EMB_F32);
  float* inv_u = (float*)alloc(NROWS * sizeof(float));
  float* inv_i = (float*)alloc(NROWS * sizeof(float));
  float* part  = (float*)alloc(256 * sizeof(float));
  float* meanv = (float*)alloc(256);

  dim3 b256(256);
  const int nElem = NROWS * DDIM;                 // 524288
  const int gElem = (nElem + 255) / 256;          // 2048

  init_side_kernel<<<gElem, b256, 0, stream>>>(emb_u, eu16, uT, u_sum);
  init_side_kernel<<<gElem, b256, 0, stream>>>(emb_i, ei16, iT, i_sum);

  // 512*512 tiles, 8 waves/block -> 32768 blocks
  compute_a_kernel<<<32768, b256, 0, stream>>>(eu16, ei16, adj, a_h, at_h);

  for (int l = 0; l < 3; ++l) {
    layer_gemm_kernel<<<64, b256, 0, stream>>>(a_h,  iT, u_new);  // u' = a  @ items
    layer_gemm_kernel<<<64, b256, 0, stream>>>(at_h, uT, i_new);  // i' = aT @ users
    postlayer_kernel<<<gElem, b256, 0, stream>>>(u_new, u_sum, uT);
    postlayer_kernel<<<gElem, b256, 0, stream>>>(i_new, i_sum, iT);
  }

  const float scale = 1.0f / (float)nElem;
  // users side (u_new holds last layer)
  reduce1_kernel<<<256, b256, 0, stream>>>(u_new, part, nElem);
  reduce2_kernel<<<1, b256, 0, stream>>>(part, meanv, 256, scale);
  center_norm_kernel<<<1024, b256, 0, stream>>>(u_new, meanv, xc_u, inv_u);
  // items side (reuse part/meanv; stream-ordered)
  reduce1_kernel<<<256, b256, 0, stream>>>(i_new, part, nElem);
  reduce2_kernel<<<1, b256, 0, stream>>>(part, meanv, 256, scale);
  center_norm_kernel<<<1024, b256, 0, stream>>>(i_new, meanv, xc_i, inv_i);

  topk_final_kernel<<<1024, b256, 0, stream>>>(xc_u, inv_u, emb_u, u_sum, out_users);
  topk_final_kernel<<<1024, b256, 0, stream>>>(xc_i, inv_i, emb_i, i_sum, out_items);
}
